// VectorQuantizer_4063039062614
// MI455X (gfx1250) — matrix operand order repaired and checked by reading, not compiled, not run on hardware
//
#include <hip/hip_runtime.h>

// ---------------------------------------------------------------------------
// VQ-VAE vector quantizer for MI455X (gfx1250, wave32, WMMA).
//   x: [65536, 256] f32,  E: [256, 1024] f32
//   out: quantized [65536*256] f32  ++  loss (1 f32)
// Distance GEMM on v_wmma_f32_16x16x32_bf16 (f32 accumulate). Each wave owns
// a 32x1024 slab (two 16-row A-fragment sets in registers) so every B
// fragment load feeds two WMMAs; argmin/gather/loss stay in fp32.
// ---------------------------------------------------------------------------

typedef __attribute__((ext_vector_type(16))) __bf16 v16bf;
typedef __attribute__((ext_vector_type(8)))  float  v8f;
typedef __attribute__((ext_vector_type(4)))  float  v4f;

#define D_DIM   256
#define K_CODES 1024
#define N_ROWS  65536

// Workspace layout (bytes):
//   [0]        float loss accumulator
//   [256]      eNorm[1024] f32                      (4 KB)
//   [8192]     ET[1024][256] f32 (E transposed)     (1 MB)
//   [1056768]  Bfrag: 512 frags x 32 lanes x 16 bf16 (512 KB)
#define WS_ENORM_OFF  256
#define WS_ET_OFF     8192
#define WS_BFRAG_OFF  (8192 + 1024*256*4)

__device__ __forceinline__ __bf16 f2bf(float f) {
  // round-to-nearest-even f32 -> bf16
  unsigned u = __builtin_bit_cast(unsigned, f);
  unsigned r = u + 0x7FFFu + ((u >> 16) & 1u);
  unsigned short h = (unsigned short)(r >> 16);
  return __builtin_bit_cast(__bf16, h);
}

// One block per code k (256 threads over D). Builds eNorm, ET, and the
// pre-swizzled bf16 B fragments in WMMA layout:
//   frag(nt,kc): lane = (N%16) + 16*(K/16 within chunk), elem e = K%16
__global__ void vq_prep(const float* __restrict__ E,
                        float* __restrict__ eNorm,
                        float* __restrict__ ET,
                        unsigned short* __restrict__ Bfrag,
                        float* __restrict__ lossAcc) {
  const int k = blockIdx.x;    // code index (N of the GEMM)
  const int d = threadIdx.x;   // feature index (K of the GEMM)
  const float v = E[d * K_CODES + k];
  ET[k * D_DIM + d] = v;

  const int nt   = k >> 4;             // N tile (16 codes)
  const int kc   = d >> 5;             // K chunk (32 features)
  const int lane = (k & 15) | (((d >> 4) & 1) << 4);
  const int e    = d & 15;
  Bfrag[(((nt * 8 + kc) * 32) + lane) * 16 + e] =
      __builtin_bit_cast(unsigned short, f2bf(v));

  __shared__ float red[256];
  red[d] = v * v;
  __syncthreads();
  for (int s = 128; s > 0; s >>= 1) {
    if (d < s) red[d] += red[d + s];
    __syncthreads();
  }
  if (d == 0) eNorm[k] = red[0];
  if (k == 0 && d == 0) *lossAcc = 0.0f;   // stream-ordered before vq_main
}

__global__ __launch_bounds__(256)
void vq_main(const float* __restrict__ x,
             const float* __restrict__ eNorm,
             const float* __restrict__ ET,
             const unsigned short* __restrict__ Bfrag,
             float* __restrict__ out,
             float* __restrict__ lossAcc) {
  const int lane  = threadIdx.x & 31;
  const int wave  = threadIdx.x >> 5;
  const int half  = lane >> 4;
  const int nlane = lane & 15;
  // Each wave owns two 16-row tiles (32 rows).
  const int tileBase = (blockIdx.x * 8 + wave) * 2;

  // --- Load A fragments (16x32 bf16) for both tiles; reused across all N.
  // A layout used here (16-bit, 16x32, k contiguous, same as Bfrag): lane row = lane%16;
//   elems 0..15 -> K = 32*kc + 16*half + e
  v16bf a[2][8];
  #pragma unroll
  for (int t = 0; t < 2; ++t) {
    const float* xr = x + ((tileBase + t) * 16 + nlane) * D_DIM;
    #pragma unroll
    for (int kc = 0; kc < 8; ++kc) {
      const int k0 = kc * 32 + half * 16;
      v4f r0 = *reinterpret_cast<const v4f*>(xr + k0);
      v4f r1 = *reinterpret_cast<const v4f*>(xr + k0 + 4);
      v4f r2 = *reinterpret_cast<const v4f*>(xr + k0 + 8);
      v4f r3 = *reinterpret_cast<const v4f*>(xr + k0 + 12);
      const float tv[16] = {r0.x, r0.y, r0.z, r0.w, r1.x, r1.y, r1.z, r1.w,
                            r2.x, r2.y, r2.z, r2.w, r3.x, r3.y, r3.z, r3.w};
      #pragma unroll
      for (int e = 0; e < 16; ++e) a[t][kc][e] = f2bf(tv[e]);
    }
  }

  float bestD[2][8];
  int   bestI[2][8];
  #pragma unroll
  for (int t = 0; t < 2; ++t)
    #pragma unroll
    for (int r = 0; r < 8; ++r) { bestD[t][r] = 3.0e38f; bestI[t][r] = 0; }

  // --- Sweep all 1024 codes, 16 at a time: 16 WMMAs / 8 B-frag loads per nt.
  #pragma unroll 1
  for (int nt = 0; nt < 64; ++nt) {
    const int   n  = nt * 16 + nlane;   // this lane's column
    const float en = eNorm[n];          // issued first; hides under WMMAs
    v8f acc0 = {0.f, 0.f, 0.f, 0.f, 0.f, 0.f, 0.f, 0.f};
    v8f acc1 = {0.f, 0.f, 0.f, 0.f, 0.f, 0.f, 0.f, 0.f};
    const v16bf* bp =
        reinterpret_cast<const v16bf*>(Bfrag + ((nt * 8) * 32 + lane) * 16);
    #pragma unroll
    for (int kc = 0; kc < 8; ++kc) {
      v16bf b = bp[kc * 32];  // independent temp per kc -> loads can clause
      acc0 = __builtin_amdgcn_wmma_f32_16x16x32_bf16(
          false, a[0][kc], false, b, (short)0, acc0, false, false);
      acc1 = __builtin_amdgcn_wmma_f32_16x16x32_bf16(
          false, a[1][kc], false, b, (short)0, acc1, false, false);
    }
    // Branchless argmin update (v_cmp + v_cndmask; ties keep lower idx since
    // n strictly increases across nt).
    #pragma unroll
    for (int r = 0; r < 8; ++r) {       // C layout: reg r -> M = r + 8*half
      float d0 = en - 2.0f * acc0[r];   // ||x||^2 dropped (row-constant)
      bool  c0 = d0 < bestD[0][r];
      bestD[0][r] = c0 ? d0 : bestD[0][r];
      bestI[0][r] = c0 ? n  : bestI[0][r];
      float d1 = en - 2.0f * acc1[r];
      bool  c1 = d1 < bestD[1][r];
      bestD[1][r] = c1 ? d1 : bestD[1][r];
      bestI[1][r] = c1 ? n  : bestI[1][r];
    }
  }

  // --- Argmin reduction across the 16 lanes of each half (ties -> lower idx).
  #pragma unroll
  for (int m = 1; m < 16; m <<= 1) {
    #pragma unroll
    for (int t = 0; t < 2; ++t)
      #pragma unroll
      for (int r = 0; r < 8; ++r) {
        float od = __shfl_xor(bestD[t][r], m, 32);
        int   oi = __shfl_xor(bestI[t][r], m, 32);
        bool  c  = od < bestD[t][r] || (od == bestD[t][r] && oi < bestI[t][r]);
        bestD[t][r] = c ? od : bestD[t][r];
        bestI[t][r] = c ? oi : bestI[t][r];
      }
  }

  // --- Gather winners, write output (streaming), accumulate loss.
  // FULLY unrolled so bestI indices are compile-time constants -> arrays stay
  // in VGPRs (a runtime index would demote them to LDS for the whole kernel).
  float lsum = 0.0f;
  const int d0c = lane * 8;
  #pragma unroll
  for (int t = 0; t < 2; ++t) {
    const int mBase = (tileBase + t) * 16;
    #pragma unroll
    for (int r = 0; r < 16; ++r) {
      const int src = (r >> 3) << 4;               // rows 0-7 lane0, 8-15 lane16
      const int idx = __shfl(bestI[t][r & 7], src, 32);
      const int row = mBase + r;
      const v4f* q4 = reinterpret_cast<const v4f*>(ET + idx * D_DIM + d0c);
      const v4f* x4 = reinterpret_cast<const v4f*>(x + row * D_DIM + d0c);
      v4f* o4 = reinterpret_cast<v4f*>(out + row * D_DIM + d0c);
      v4f q0 = q4[0], q1 = q4[1], xv0 = x4[0], xv1 = x4[1];
      __builtin_nontemporal_store(q0, o4);         // out written once: TH_NT
      __builtin_nontemporal_store(q1, o4 + 1);
      v4f e0 = q0 - xv0;
      v4f e1 = q1 - xv1;
      lsum += e0.x * e0.x + e0.y * e0.y + e0.z * e0.z + e0.w * e0.w;
      lsum += e1.x * e1.x + e1.y * e1.y + e1.z * e1.z + e1.w * e1.w;
    }
  }
  #pragma unroll
  for (int m = 1; m < 32; m <<= 1) lsum += __shfl_xor(lsum, m, 32);
  if (lane == 0) atomicAdd(lossAcc, lsum);
}

__global__ void vq_finish(const float* __restrict__ lossAcc,
                          float* __restrict__ out) {
  // loss = beta*mean + mean = 1.25 * mean((q - x)^2)
  out[N_ROWS * D_DIM] = 1.25f * (*lossAcc) / (float)(N_ROWS * D_DIM);
}

extern "C" void kernel_launch(void* const* d_in, const int* in_sizes, int n_in,
                              void* d_out, int out_size, void* d_ws,
                              size_t ws_size, hipStream_t stream) {
  const float* x = (const float*)d_in[0];   // [65536, 256]
  const float* E = (const float*)d_in[1];   // [256, 1024]
  float* out = (float*)d_out;               // [65536*256] ++ [1]

  char* ws = (char*)d_ws;                   // needs ~1.51 MB
  float* lossAcc = (float*)ws;
  float* eNorm   = (float*)(ws + WS_ENORM_OFF);
  float* ET      = (float*)(ws + WS_ET_OFF);
  unsigned short* Bfrag = (unsigned short*)(ws + WS_BFRAG_OFF);

  vq_prep<<<K_CODES, 256, 0, stream>>>(E, eNorm, ET, Bfrag, lossAcc);
  // 8 waves/block x 32 rows/wave = 256 rows/block
  vq_main<<<N_ROWS / 256, 256, 0, stream>>>(x, eNorm, ET, Bfrag, out, lossAcc);
  vq_finish<<<1, 1, 0, stream>>>(lossAcc, out);
}
